// MultiLayerBTreeLSTM_13615046329043
// MI455X (gfx1250) — matrix operand order repaired and checked by reading, not compiled, not run on hardware
//
#include <hip/hip_runtime.h>
#include <hip/hip_bf16.h>

typedef __attribute__((ext_vector_type(16))) __bf16 v16bf;
typedef __attribute__((ext_vector_type(8)))  __bf16 v8bf;
typedef __attribute__((ext_vector_type(8)))  float  v8f;

#define TREE_D 18
#define NNODES ((1 << TREE_D) - 1)
#define HDIM 64

__device__ __forceinline__ unsigned short f2bf(float f) {
    union { float f; unsigned u; } v; v.f = f;
    unsigned u = v.u;
    u += 0x7FFFu + ((u >> 16) & 1u);   // round-to-nearest-even
    return (unsigned short)(u >> 16);
}
__device__ __forceinline__ float sigf(float x) { return 1.0f / (1.0f + __expf(-x)); }
__device__ __forceinline__ float tanh_fast(float x) {
    float a = __expf(-2.0f * fabsf(x));
    float t = (1.0f - a) / (1.0f + a);
    return copysignf(t, x);
}

// ---------------------------------------------------------------------------
// Pack fwd weights into bf16 concat matrix (512 rows, rows 448..511 zero-pad):
// rows 0..383 = [Wx | Wl | Wr], rows 384..447 = [Wpx | 0 | 0], 448.. = 0.
// bias[512] = [bx+bl+br | bpx | 0].  Row stride = KT = in_d + 128.
// ---------------------------------------------------------------------------
__global__ void build_up_w(const float* __restrict__ Wx, const float* __restrict__ Wl,
                           const float* __restrict__ Wr, const float* __restrict__ Wpx,
                           const float* __restrict__ bx, const float* __restrict__ bl,
                           const float* __restrict__ br, const float* __restrict__ bpx,
                           int in_d, unsigned short* __restrict__ Wout,
                           float* __restrict__ bout) {
    const int KT = in_d + 128;
    const int total = 512 * KT;
    const int stride = gridDim.x * blockDim.x;
    for (int idx = blockIdx.x * blockDim.x + threadIdx.x; idx < total; idx += stride) {
        int row = idx / KT, k = idx - row * KT;
        float v = 0.0f;
        if (row < 384) {
            if (k < in_d)            v = Wx[row * in_d + k];
            else if (k < in_d + 64)  v = Wl[row * 64 + (k - in_d)];
            else                     v = Wr[row * 64 + (k - in_d - 64)];
        } else if (row < 448) {
            int pr = row - 384;
            if (k < in_d)            v = Wpx[pr * in_d + k];
        }
        Wout[idx] = f2bf(v);
    }
    for (int r = blockIdx.x * blockDim.x + threadIdx.x; r < 512; r += stride)
        bout[r] = (r < 384) ? (bx[r] + bl[r] + br[r]) : (r < 448 ? bpx[r - 384] : 0.0f);
}

// ---------------------------------------------------------------------------
// Pack bwd weights: rows 0..319 = [Wx | Wh], rows 320..383 = [Wpx | 0];
// bias[384] = [bx+bh | bpx].  Row stride = KT = in_d + 64.
// ---------------------------------------------------------------------------
__global__ void build_dn_w(const float* __restrict__ Wx, const float* __restrict__ Wh,
                           const float* __restrict__ Wpx,
                           const float* __restrict__ bx, const float* __restrict__ bh,
                           const float* __restrict__ bpx,
                           int in_d, unsigned short* __restrict__ Wout,
                           float* __restrict__ bout) {
    const int KT = in_d + 64;
    const int total = 384 * KT;
    const int stride = gridDim.x * blockDim.x;
    for (int idx = blockIdx.x * blockDim.x + threadIdx.x; idx < total; idx += stride) {
        int row = idx / KT, k = idx - row * KT;
        float v = 0.0f;
        if (row < 320) {
            if (k < in_d) v = Wx[row * in_d + k];
            else          v = Wh[row * 64 + (k - in_d)];
        } else {
            int pr = row - 320;
            if (k < in_d) v = Wpx[pr * in_d + k];
        }
        Wout[idx] = f2bf(v);
    }
    for (int r = blockIdx.x * blockDim.x + threadIdx.x; r < 384; r += stride)
        bout[r] = (r < 320) ? (bx[r] + bh[r]) : bpx[r - 320];
}

// ---------------------------------------------------------------------------
// One tree level. Block = 256 threads = 8 wave32s, one block per 16-node tile.
//   g[16 x NTOT] = bf16([x | x1 | h_nbrs])[16 x KTOT] @ Wc^T + bias  (WMMA)
// then the LSTM cell elementwise.  CTW (compile-time) column tiles per wave:
// every A fragment load feeds CTW WMMAs with no divergence / no acc copies.
// ---------------------------------------------------------------------------
template <int CTW>
__global__ __launch_bounds__(256)
void tree_sweep(const float* __restrict__ x0, int w0,
                const float* __restrict__ x1, int w1,
                const unsigned short* __restrict__ Wc,
                const float* __restrict__ bias,
                float* __restrict__ h_out, int h_stride,
                float* __restrict__ c_arr,
                int n, int off, int KTOT, int NTOT,
                int isUp, int hasChild) {
    extern __shared__ char smem[];
    const int LDA = KTOT + 8;            // bf16 elems, padded: conflict-free ds_load_b128
    const int GLD = NTOT + 4;            // f32 elems
    unsigned short* ldsA = (unsigned short*)smem;
    float* gf = (float*)(smem + (size_t)16 * LDA * sizeof(unsigned short));

    const int tid   = threadIdx.x;
    const int tile0 = blockIdx.x * 16;   // first node (within level) of this tile

    // ---- Stage A = [x | neighbor h] into LDS as bf16 (float4 / b64 packed) ----
    const int KT4 = KTOT >> 2;
    for (int idx = tid; idx < 16 * KT4; idx += 256) {
        int r = idx / KT4;
        int k = (idx - r * KT4) << 2;    // all segments are 64-aligned, k%4==0 safe
        int m = tile0 + r;
        float4 v = make_float4(0.f, 0.f, 0.f, 0.f);
        if (m < n) {
            int gid = off + m;
            if (k < w0) {
                v = *(const float4*)(x0 + (size_t)gid * w0 + k);
            } else if (k < w0 + w1) {
                v = *(const float4*)(x1 + (size_t)gid * w1 + (k - w0));
            } else {
                int kk = k - w0 - w1;
                if (isUp) {
                    if (hasChild) {
                        int which = kk >> 6;           // 0 = left child h, 1 = right
                        int j = kk & 63;
                        int gc = 2 * gid + 1 + which;
                        v = *(const float4*)(h_out + (size_t)gc * h_stride + j);
                    }
                } else if (gid > 0) {
                    int gp = (gid - 1) >> 1;
                    v = *(const float4*)(h_out + (size_t)gp * h_stride + kk);
                }
            }
        }
        unsigned long long pk =  (unsigned long long)f2bf(v.x)
                              | ((unsigned long long)f2bf(v.y) << 16)
                              | ((unsigned long long)f2bf(v.z) << 32)
                              | ((unsigned long long)f2bf(v.w) << 48);
        *(unsigned long long*)(ldsA + r * LDA + k) = pk;   // ds_store_b64
    }
    __syncthreads();

    // ---- WMMA: wave owns column tiles wid, wid+8, ..., wid+8*(CTW-1) ----
    const int lane = tid & 31;
    const int wid  = tid >> 5;
    const int lrow = lane & 15;
    const int hi   = lane >> 4;          // A, B: K 0-7/16-23 vs 8-15/24-31
    const int KS = KTOT >> 5;

    const unsigned short* wp[CTW];
#pragma unroll
    for (int j = 0; j < CTW; ++j)
        wp[j] = Wc + (size_t)((wid + 8 * j) * 16 + lrow) * KTOT;

    v8f acc[CTW];
#pragma unroll
    for (int j = 0; j < CTW; ++j) acc[j] = (v8f){0.f, 0.f, 0.f, 0.f, 0.f, 0.f, 0.f, 0.f};

    for (int ks = 0; ks < KS; ++ks) {
        const unsigned short* pa = ldsA + lrow * LDA + ks * 32 + hi * 8;
        v8bf alo = *(const v8bf*)pa;          // ds_load_b128
        v8bf ahi = *(const v8bf*)(pa + 16);
        v16bf a = __builtin_shufflevector(alo, ahi, 0,1,2,3,4,5,6,7,8,9,10,11,12,13,14,15);
#pragma unroll
        for (int j = 0; j < CTW; ++j) {
            const unsigned short* pb = wp[j] + ks * 32 + hi * 8;
            v8bf blo = *(const v8bf*)pb;      // global_load_b128 (L0-resident weights)
            v8bf bhi = *(const v8bf*)(pb + 16);
            v16bf b = __builtin_shufflevector(blo, bhi, 0,1,2,3,4,5,6,7,8,9,10,11,12,13,14,15);
            acc[j] = __builtin_amdgcn_wmma_f32_16x16x32_bf16(
                false, a, false, b, (short)0, acc[j], false, false);
        }
    }

#pragma unroll
    for (int j = 0; j < CTW; ++j) {
        const int ct = wid + 8 * j;
        const float bc = bias[ct * 16 + lrow];
#pragma unroll
        for (int r = 0; r < 8; ++r)
            gf[(r + 8 * hi) * GLD + ct * 16 + lrow] = acc[j][r] + bc;
    }
    __syncthreads();

    // ---- Elementwise LSTM cell: 16 nodes x 64 h-elems ----
    for (int e = tid; e < 16 * HDIM; e += 256) {
        int r = e >> 6;
        int j = e & 63;
        int m = tile0 + r;
        if (m >= n) continue;
        int gid = off + m;
        const float* grow = gf + r * GLD;
        float c, px, rr, go;
        if (isUp) {
            float gi = grow[j],        goo = grow[64 + j];
            float fl = grow[128 + j],  fr  = grow[192 + j];
            float gu = grow[256 + j];  rr  = grow[320 + j];
            px = grow[384 + j];
            float lc = 0.0f, rc = 0.0f;
            if (hasChild) {
                lc = c_arr[(size_t)(2 * gid + 1) * HDIM + j];
                rc = c_arr[(size_t)(2 * gid + 2) * HDIM + j];
            }
            c = sigf(gi) * tanh_fast(gu) + sigf(fl) * lc + sigf(fr) * rc;
            go = goo;
        } else {
            float gi = grow[j],        goo = grow[64 + j];
            float gff = grow[128 + j], gu  = grow[192 + j];
            rr = grow[256 + j];
            px = grow[320 + j];
            float pc = (gid > 0) ? c_arr[(size_t)((gid - 1) >> 1) * HDIM + j] : 0.0f;
            c = sigf(gi) * tanh_fast(gu) + sigf(gff) * pc;
            go = goo;
        }
        float h  = sigf(go) * tanh_fast(c);
        float rv = sigf(rr);
        c_arr[(size_t)gid * HDIM + j] = c;
        h_out[(size_t)gid * h_stride + j] = rv * h + (1.0f - rv) * px;
    }
}

// ---------------------------------------------------------------------------
// Host driver
// ---------------------------------------------------------------------------
extern "C" void kernel_launch(void* const* d_in, const int* in_sizes, int n_in,
                              void* d_out, int out_size, void* d_ws, size_t ws_size,
                              hipStream_t stream) {
    (void)in_sizes; (void)n_in; (void)out_size; (void)ws_size;
    // JAX pytree flatten order: 'features', then params list; each layer dict
    // sorts 'bwd' < 'fwd'; bwd keys: Wh,Wpx,Wx,bh,bpx,bx; fwd: Wl,Wpx,Wr,Wx,bl,bpx,br,bx.
    const float* feat = (const float*)d_in[0];
    struct BW { const float *Wh, *Wpx, *Wx, *bh, *bpx, *bx; };
    struct FW { const float *Wl, *Wpx, *Wr, *Wx, *bl, *bpx, *br, *bx; };
    BW bw[2]; FW fw[2];
    int p = 1;
    for (int L = 0; L < 2; ++L) {
        bw[L].Wh  = (const float*)d_in[p++]; bw[L].Wpx = (const float*)d_in[p++];
        bw[L].Wx  = (const float*)d_in[p++]; bw[L].bh  = (const float*)d_in[p++];
        bw[L].bpx = (const float*)d_in[p++]; bw[L].bx  = (const float*)d_in[p++];
        fw[L].Wl  = (const float*)d_in[p++]; fw[L].Wpx = (const float*)d_in[p++];
        fw[L].Wr  = (const float*)d_in[p++]; fw[L].Wx  = (const float*)d_in[p++];
        fw[L].bl  = (const float*)d_in[p++]; fw[L].bpx = (const float*)d_in[p++];
        fw[L].br  = (const float*)d_in[p++]; fw[L].bx  = (const float*)d_in[p++];
    }

    auto au = [](size_t x) { return (x + 255) & ~(size_t)255; };
    char* ws = (char*)d_ws;
    size_t o = 0;
    const size_t szH = (size_t)NNODES * HDIM * sizeof(float);
    float* hU0 = (float*)(ws + o); o = au(o + szH);
    float* hD0 = (float*)(ws + o); o = au(o + szH);
    float* cb  = (float*)(ws + o); o = au(o + szH);
    unsigned short* wup[2]; unsigned short* wdn[2];
    float* bup[2]; float* bdn[2];
    const int in_d[2] = {256, 128};
    for (int L = 0; L < 2; ++L) {
        wup[L] = (unsigned short*)(ws + o); o = au(o + (size_t)512 * (in_d[L] + 128) * 2);
        bup[L] = (float*)(ws + o);          o = au(o + 512 * sizeof(float));
        wdn[L] = (unsigned short*)(ws + o); o = au(o + (size_t)384 * (in_d[L] + 64) * 2);
        bdn[L] = (float*)(ws + o);          o = au(o + 384 * sizeof(float));
    }

    // Pack weights to bf16 once per call.
    for (int L = 0; L < 2; ++L) {
        int gu = (512 * (in_d[L] + 128) + 255) / 256;
        build_up_w<<<gu, 256, 0, stream>>>(fw[L].Wx, fw[L].Wl, fw[L].Wr, fw[L].Wpx,
                                           fw[L].bx, fw[L].bl, fw[L].br, fw[L].bpx,
                                           in_d[L], wup[L], bup[L]);
        int gd = (384 * (in_d[L] + 64) + 255) / 256;
        build_dn_w<<<gd, 256, 0, stream>>>(bw[L].Wx, bw[L].Wh, bw[L].Wpx,
                                           bw[L].bx, bw[L].bh, bw[L].bpx,
                                           in_d[L], wdn[L], bdn[L]);
    }

    float* out = (float*)d_out;
    auto shmem = [](int KT, int NT) {
        return (size_t)16 * (KT + 8) * 2 + (size_t)16 * (NT + 4) * 4;
    };

    for (int L = 0; L < 2; ++L) {
        const float* x0; const float* x1; int w0, w1;
        float* hu; float* hd; int hs;
        if (L == 0) { x0 = feat; w0 = 256; x1 = nullptr; w1 = 0; hu = hU0; hd = hD0; hs = 64; }
        else        { x0 = hU0;  w0 = 64;  x1 = hD0;     w1 = 64; hu = out; hd = out + 64; hs = 128; }
        const int KTu = w0 + w1 + 128, NTu = 512;   // padded: 32 col tiles, 4/wave
        const int KTd = w0 + w1 + 64,  NTd = 384;   // 24 col tiles, 3/wave
        // up sweep: leaves -> root
        for (int l = TREE_D - 1; l >= 0; --l) {
            int n = 1 << l, off = n - 1, tiles = (n + 15) / 16;
            tree_sweep<4><<<tiles, 256, shmem(KTu, NTu), stream>>>(
                x0, w0, x1, w1, wup[L], bup[L], hu, hs, cb,
                n, off, KTu, NTu, /*isUp=*/1, /*hasChild=*/(l != TREE_D - 1));
        }
        // down sweep: root -> leaves
        for (int l = 0; l < TREE_D; ++l) {
            int n = 1 << l, off = n - 1, tiles = (n + 15) / 16;
            tree_sweep<3><<<tiles, 256, shmem(KTd, NTd), stream>>>(
                x0, w0, x1, w1, wdn[L], bdn[L], hd, hs, cb,
                n, off, KTd, NTd, /*isUp=*/0, /*hasChild=*/0);
        }
    }
}
